// SpatialGNNEncoder_35390530519600
// MI455X (gfx1250) — compile-verified, matrix operand order checked by reading, not run on hardware
//
#include <hip/hip_runtime.h>
#include <math.h>

// ---------------- problem constants ----------------
constexpr int NN = 50000;   // nodes  (multiple of 16)
constexpr int NE = 800000;  // edges  (multiple of 16)
constexpr int NG = 64;      // graphs
constexpr int IN = 64;
constexpr int ED = 16;
constexpr int HID = 32;
constexpr int H1 = 4;
constexpr int W1 = 128;     // HID*H1
constexpr float EPS = 1e-5f;
constexpr float INV_SQRT_D = 0.17677669529663687f; // 1/sqrt(32)

typedef __attribute__((ext_vector_type(16))) _Float16 v16h;
typedef __attribute__((ext_vector_type(8)))  float    v8f;

// ---------------- small device helpers ----------------
__device__ __forceinline__ float wave_reduce_sum(float t) {
  #pragma unroll
  for (int off = 16; off > 0; off >>= 1) t += __shfl_xor(t, off, 32);
  return t; // all lanes hold the sum (butterfly)
}

__device__ __forceinline__ void atomicMaxF(float* addr, float val) {
  // monotone float max via int/uint atomics (init must be -INF)
  if (val >= 0.0f) atomicMax((int*)addr, __float_as_int(val));
  else             atomicMin((unsigned int*)addr, __float_as_uint(val));
}

// ---------------- init / fill ----------------
__global__ void fill_f32(float* p, float v, int n) {
  int i = blockIdx.x * blockDim.x + threadIdx.x;
  if (i < n) p[i] = v;
}

// ---------------- BatchNorm (eval) ----------------
__global__ void bn_to_f16(const float* __restrict__ x, const float* __restrict__ g,
                          const float* __restrict__ b, const float* __restrict__ m,
                          const float* __restrict__ v, _Float16* __restrict__ out,
                          int total, int C) {
  int i = blockIdx.x * blockDim.x + threadIdx.x;
  if (i >= total) return;
  int c = i % C;
  out[i] = (_Float16)((x[i] - m[c]) * rsqrtf(v[c] + EPS) * g[c] + b[c]);
}

__global__ void bn_to_f32(const float* __restrict__ x, const float* __restrict__ g,
                          const float* __restrict__ b, const float* __restrict__ m,
                          const float* __restrict__ v, float* __restrict__ out,
                          int total, int C) {
  int i = blockIdx.x * blockDim.x + threadIdx.x;
  if (i >= total) return;
  int c = i % C;
  out[i] = (x[i] - m[c]) * rsqrtf(v[c] + EPS) * g[c] + b[c];
}

// ---------------- weight packing into WMMA B-fragment order ----------------
// Bp[((kt*nTiles+nt)*32 + lane)*16 + i] = B[k, n]  (f16)
//   g = lane>>4, n = nt*16 + (lane&15)
//   k = kt*32 + (i<8 ? g*8+i : 16+g*8+(i-8))
// B = [W0 | W1 | W2 | W3] columnwise, each [K, colsPer]
__global__ void pack_weights(const float* __restrict__ W0, const float* __restrict__ W1,
                             const float* __restrict__ W2, const float* __restrict__ W3,
                             _Float16* __restrict__ Bp, int K, int colsPer) {
  int nTiles = (4 * colsPer) >> 4;
  int total = (K >> 5) * nTiles * 32 * 16;
  int tid = blockIdx.x * blockDim.x + threadIdx.x;
  if (tid >= total) return;
  int i    = tid & 15;
  int lane = (tid >> 4) & 31;
  int t    = tid >> 9;
  int nt = t % nTiles;
  int kt = t / nTiles;
  int g = lane >> 4;
  int k = kt * 32 + ((i < 8) ? (g * 8 + i) : (16 + g * 8 + (i - 8)));
  int n = nt * 16 + (lane & 15);
  int sel = n / colsPer;
  const float* W = (sel == 0) ? W0 : (sel == 1) ? W1 : (sel == 2) ? W2 : W3;
  Bp[tid] = (_Float16)W[(size_t)k * colsPer + (n % colsPer)];
}

__global__ void pack_bias(const float* __restrict__ b0, const float* __restrict__ b1,
                          const float* __restrict__ b2, const float* __restrict__ b3,
                          float* __restrict__ bc, int colsPer) {
  int n = blockIdx.x * blockDim.x + threadIdx.x;
  if (n >= 4 * colsPer) return;
  int sel = n / colsPer;
  const float* b = (sel == 0) ? b0 : (sel == 1) ? b1 : (sel == 2) ? b2 : b3;
  bc[n] = b[n % colsPer];
}

// ---------------- WMMA GEMM: C[M,Nc] = f32( A_f16[M,K] @ Bpacked_f16[K,Nc] ) + bias ----------------
// one wave32 per 16x16 output tile, K stepped by 32 (v_wmma_f32_16x16x32_f16)
__global__ void gemm_wmma_f16(const _Float16* __restrict__ A, const _Float16* __restrict__ Bp,
                              const float* __restrict__ bias, float* __restrict__ C,
                              int M, int K, int Nc) {
  const int lane = threadIdx.x & 31;
  const int wid  = (blockIdx.x * blockDim.x + threadIdx.x) >> 5;
  const int nTiles = Nc >> 4;
  const int mt = wid / nTiles;
  const int nt = wid - mt * nTiles;
  if (mt * 16 >= M) return;
  const int g   = lane >> 4;
  const int row = mt * 16 + (lane & 15);
  v8f acc = {0.f, 0.f, 0.f, 0.f, 0.f, 0.f, 0.f, 0.f};
  const int kTiles = K >> 5;
  for (int kt = 0; kt < kTiles; ++kt) {
    union { float4 f[2]; v16h v; } a, b;
    const _Float16* ap = A + (size_t)row * K + kt * 32 + g * 8;
    a.f[0] = *reinterpret_cast<const float4*>(ap);       // K = kt*32 + g*8 .. +7
    a.f[1] = *reinterpret_cast<const float4*>(ap + 16);  // K = kt*32 + 16 + g*8 .. +7
    const _Float16* bp = Bp + (((size_t)kt * nTiles + nt) * 32 + lane) * 16;
    b.f[0] = *reinterpret_cast<const float4*>(bp);
    b.f[1] = *reinterpret_cast<const float4*>(bp + 8);
    acc = __builtin_amdgcn_wmma_f32_16x16x32_f16(false, a.v, false, b.v,
                                                 (short)0, acc, false, false);
  }
  const int n  = nt * 16 + (lane & 15);
  const float bv = bias[n];
  #pragma unroll
  for (int r = 0; r < 8; ++r) {
    const int m = mt * 16 + g * 8 + r;
    C[(size_t)m * Nc + n] = acc[r] + bv;
  }
}

// ---------------- edge attention (wave per edge, lane = d in 0..31) ----------------
__global__ void edge_logits(const int* __restrict__ ei, const float* __restrict__ ea,
                            const float* __restrict__ QKVS, int qkvsStride, int kOff,
                            const float* __restrict__ We, int weCols, int heads,
                            float* __restrict__ aOut, float* __restrict__ mOut) {
  int wid  = (blockIdx.x * blockDim.x + threadIdx.x) >> 5;
  int lane = threadIdx.x & 31;
  if (wid >= NE) return;
  int src = ei[wid], dst = ei[NE + wid];
  for (int h = 0; h < heads; ++h) {
    float eh = 0.f;
    #pragma unroll
    for (int j = 0; j < ED; ++j)
      eh += ea[(size_t)wid * ED + j] * We[j * weCols + h * 32 + lane];
    float kv = QKVS[(size_t)src * qkvsStride + kOff + h * 32 + lane] + eh;
    float qv = QKVS[(size_t)dst * qkvsStride + h * 32 + lane];
    float t = wave_reduce_sum(qv * kv) * INV_SQRT_D;
    if (lane == 0) {
      aOut[(size_t)wid * heads + h] = t;
      atomicMaxF(&mOut[(size_t)dst * heads + h], t);
    }
  }
}

__global__ void edge_soft(const int* __restrict__ ei, float* __restrict__ a,
                          const float* __restrict__ m, float* __restrict__ s, int heads) {
  int t = blockIdx.x * blockDim.x + threadIdx.x;
  if (t >= NE * heads) return;
  int e = t / heads, h = t - e * heads;
  int dst = ei[NE + e];
  float p = expf(a[t] - m[(size_t)dst * heads + h]);
  a[t] = p;
  atomicAdd(&s[(size_t)dst * heads + h], p);
}

__global__ void edge_msg(const int* __restrict__ ei, const float* __restrict__ ea,
                         const float* __restrict__ QKVS, int qkvsStride, int vOff,
                         const float* __restrict__ We, int weCols, int heads,
                         const float* __restrict__ a, const float* __restrict__ s,
                         float* __restrict__ agg, int aggStride) {
  int wid  = (blockIdx.x * blockDim.x + threadIdx.x) >> 5;
  int lane = threadIdx.x & 31;
  if (wid >= NE) return;
  int src = ei[wid], dst = ei[NE + wid];
  for (int h = 0; h < heads; ++h) {
    float eh = 0.f;
    #pragma unroll
    for (int j = 0; j < ED; ++j)
      eh += ea[(size_t)wid * ED + j] * We[j * weCols + h * 32 + lane];
    float alpha = a[(size_t)wid * heads + h] / (s[(size_t)dst * heads + h] + 1e-16f);
    float msg = (QKVS[(size_t)src * qkvsStride + vOff + h * 32 + lane] + eh) * alpha;
    atomicAdd(&agg[(size_t)dst * aggStride + h * 32 + lane], msg);
  }
}

// ---------------- beta gate + BN1 + ELU -> h1 (f16) ; wave per node, 4 dims/lane ----------------
__global__ void beta_bn_elu1(const float* __restrict__ agg, const float* __restrict__ QKVS,
                             const float* __restrict__ Wb,
                             const float* __restrict__ g1, const float* __restrict__ b1,
                             const float* __restrict__ m1, const float* __restrict__ v1,
                             _Float16* __restrict__ h1) {
  int n    = (blockIdx.x * blockDim.x + threadIdx.x) >> 5;
  int lane = threadIdx.x & 31;
  if (n >= NN) return;
  float av[4], xv[4];
  float z = 0.f;
  #pragma unroll
  for (int t = 0; t < 4; ++t) {
    int j = lane + 32 * t;
    float a_ = agg[(size_t)n * W1 + j];
    float x_ = QKVS[(size_t)n * 512 + 384 + j];
    z += a_ * Wb[j] + x_ * Wb[W1 + j] + (a_ - x_) * Wb[2 * W1 + j];
    av[t] = a_; xv[t] = x_;
  }
  z = wave_reduce_sum(z);
  float beta = 1.f / (1.f + expf(-z));
  #pragma unroll
  for (int t = 0; t < 4; ++t) {
    int j = lane + 32 * t;
    float o = beta * xv[t] + (1.f - beta) * av[t];
    float val = (o - m1[j]) * rsqrtf(v1[j] + EPS) * g1[j] + b1[j];
    val = (val > 0.f) ? val : expm1f(val);
    h1[(size_t)n * W1 + j] = (_Float16)val;
  }
}

// ---------------- beta gate + BN2 + ELU + pooling atomics ; wave per node, lane = dim ----------------
__global__ void beta_bn_elu2_pool(const float* __restrict__ agg, const float* __restrict__ QKVS,
                                  const float* __restrict__ Wb,
                                  const float* __restrict__ g2, const float* __restrict__ b2,
                                  const float* __restrict__ m2, const float* __restrict__ v2,
                                  const int* __restrict__ batch,
                                  float* __restrict__ psum, float* __restrict__ pmax,
                                  float* __restrict__ pcnt) {
  int n    = (blockIdx.x * blockDim.x + threadIdx.x) >> 5;
  int lane = threadIdx.x & 31;
  if (n >= NN) return;
  float a_ = agg[(size_t)n * HID + lane];
  float x_ = QKVS[(size_t)n * 128 + 96 + lane];
  float z = a_ * Wb[lane] + x_ * Wb[HID + lane] + (a_ - x_) * Wb[2 * HID + lane];
  z = wave_reduce_sum(z);
  float beta = 1.f / (1.f + expf(-z));
  float o = beta * x_ + (1.f - beta) * a_;
  float val = (o - m2[lane]) * rsqrtf(v2[lane] + EPS) * g2[lane] + b2[lane];
  val = (val > 0.f) ? val : expm1f(val);
  int g = batch[n];
  atomicAdd(&psum[(size_t)g * HID + lane], val);
  atomicMaxF(&pmax[(size_t)g * HID + lane], val);
  if (lane == 0) atomicAdd(&pcnt[g], 1.0f);
}

__global__ void pool_finalize(const float* __restrict__ psum, const float* __restrict__ pmax,
                              const float* __restrict__ pcnt, float* __restrict__ out) {
  int t = blockIdx.x * blockDim.x + threadIdx.x;
  if (t >= NG * 2 * HID) return;
  int g = t >> 6, d = t & 63;
  if (d < HID) {
    float c = pcnt[g]; c = (c > 1.f) ? c : 1.f;
    out[t] = psum[(size_t)g * HID + d] / c;
  } else {
    float mv = pmax[(size_t)g * HID + (d - HID)];
    out[t] = (mv < -1.0e30f) ? 0.f : mv;
  }
}

// ---------------- host launch ----------------
extern "C" void kernel_launch(void* const* d_in, const int* in_sizes, int n_in,
                              void* d_out, int out_size, void* d_ws, size_t ws_size,
                              hipStream_t stream) {
  (void)in_sizes; (void)n_in; (void)out_size; (void)ws_size;
  // input pointers (setup_inputs order)
  const float* x       = (const float*)d_in[0];
  const int*   ei      = (const int*)d_in[1];
  const float* eattr   = (const float*)d_in[2];
  const int*   batch   = (const int*)d_in[3];
  const float* bnN_g = (const float*)d_in[4],  *bnN_b = (const float*)d_in[5],
             * bnN_m = (const float*)d_in[6],  *bnN_v = (const float*)d_in[7];
  const float* bnE_g = (const float*)d_in[8],  *bnE_b = (const float*)d_in[9],
             * bnE_m = (const float*)d_in[10], *bnE_v = (const float*)d_in[11];
  const float* bn1_g = (const float*)d_in[12], *bn1_b = (const float*)d_in[13],
             * bn1_m = (const float*)d_in[14], *bn1_v = (const float*)d_in[15];
  const float* bn2_g = (const float*)d_in[16], *bn2_b = (const float*)d_in[17],
             * bn2_m = (const float*)d_in[18], *bn2_v = (const float*)d_in[19];
  const float* Wq1=(const float*)d_in[20], *bq1=(const float*)d_in[21];
  const float* Wk1=(const float*)d_in[22], *bk1=(const float*)d_in[23];
  const float* Wv1=(const float*)d_in[24], *bv1=(const float*)d_in[25];
  const float* We1=(const float*)d_in[26];
  const float* Ws1=(const float*)d_in[27], *bs1=(const float*)d_in[28];
  const float* Wb1=(const float*)d_in[29];
  const float* Wq2=(const float*)d_in[30], *bq2=(const float*)d_in[31];
  const float* Wk2=(const float*)d_in[32], *bk2=(const float*)d_in[33];
  const float* Wv2=(const float*)d_in[34], *bv2=(const float*)d_in[35];
  const float* We2=(const float*)d_in[36];
  const float* Ws2=(const float*)d_in[37], *bs2=(const float*)d_in[38];
  const float* Wb2=(const float*)d_in[39];
  float* out = (float*)d_out;

  // ---- workspace bump allocator (256B aligned) ----
  size_t off = 0;
  auto alloc = [&](size_t bytes) -> void* {
    off = (off + 255) & ~(size_t)255;
    void* p = (char*)d_ws + off;
    off += bytes;
    return p;
  };
  _Float16* h16   = (_Float16*)alloc((size_t)NN * IN * 2);
  float*    ea32  = (float*)   alloc((size_t)NE * ED * 4);
  _Float16* Wp1   = (_Float16*)alloc((size_t)2 * 32 * 32 * 16 * 2);   // K=64,Nc=512
  float*    bc1   = (float*)   alloc(512 * 4);
  _Float16* Wp2   = (_Float16*)alloc((size_t)4 * 8 * 32 * 16 * 2);    // K=128,Nc=128
  float*    bc2   = (float*)   alloc(128 * 4);
  float*    QKVS1 = (float*)   alloc((size_t)NN * 512 * 4);
  float*    a1    = (float*)   alloc((size_t)NE * H1 * 4);
  float*    mx1   = (float*)   alloc((size_t)NN * H1 * 4);
  float*    sm1   = (float*)   alloc((size_t)NN * H1 * 4);
  float*    agg1  = (float*)   alloc((size_t)NN * W1 * 4);
  _Float16* h1_16 = (_Float16*)alloc((size_t)NN * W1 * 2);
  float*    QKVS2 = (float*)   alloc((size_t)NN * 128 * 4);
  float*    a2    = (float*)   alloc((size_t)NE * 4);
  float*    mx2   = (float*)   alloc((size_t)NN * 4);
  float*    sm2   = (float*)   alloc((size_t)NN * 4);
  float*    agg2  = (float*)   alloc((size_t)NN * HID * 4);
  float*    psum  = (float*)   alloc((size_t)NG * HID * 4);
  float*    pmax  = (float*)   alloc((size_t)NG * HID * 4);
  float*    pcnt  = (float*)   alloc((size_t)NG * 4);

  const int T = 256;
  auto blocks = [](long n, int t) { return (int)((n + t - 1) / t); };

  // ---- init accumulators (must happen every call) ----
  fill_f32<<<blocks((long)NN*H1, T), T, 0, stream>>>(mx1, -INFINITY, NN*H1);
  fill_f32<<<blocks((long)NN*H1, T), T, 0, stream>>>(sm1, 0.f, NN*H1);
  fill_f32<<<blocks((long)NN*W1, T), T, 0, stream>>>(agg1, 0.f, NN*W1);
  fill_f32<<<blocks(NN, T), T, 0, stream>>>(mx2, -INFINITY, NN);
  fill_f32<<<blocks(NN, T), T, 0, stream>>>(sm2, 0.f, NN);
  fill_f32<<<blocks((long)NN*HID, T), T, 0, stream>>>(agg2, 0.f, NN*HID);
  fill_f32<<<blocks(NG*HID, T), T, 0, stream>>>(psum, 0.f, NG*HID);
  fill_f32<<<blocks(NG*HID, T), T, 0, stream>>>(pmax, -INFINITY, NG*HID);
  fill_f32<<<blocks(NG, T), T, 0, stream>>>(pcnt, 0.f, NG);

  // ---- BatchNorms ----
  bn_to_f16<<<blocks((long)NN*IN, T), T, 0, stream>>>(x, bnN_g, bnN_b, bnN_m, bnN_v, h16, NN*IN, IN);
  bn_to_f32<<<blocks((long)NE*ED, T), T, 0, stream>>>(eattr, bnE_g, bnE_b, bnE_m, bnE_v, ea32, NE*ED, ED);

  // ---- pack weights into WMMA fragment order ----
  pack_weights<<<blocks(2L*32*32*16, T), T, 0, stream>>>(Wq1, Wk1, Wv1, Ws1, Wp1, IN, W1);
  pack_bias<<<blocks(512, T), T, 0, stream>>>(bq1, bk1, bv1, bs1, bc1, W1);
  pack_weights<<<blocks(4L*8*32*16, T), T, 0, stream>>>(Wq2, Wk2, Wv2, Ws2, Wp2, W1, HID);
  pack_bias<<<blocks(128, T), T, 0, stream>>>(bq2, bk2, bv2, bs2, bc2, HID);

  // ---- conv1: projections via WMMA  [N,64] x [64,512] -> Q|K|V|S ----
  {
    long tiles = (long)(NN / 16) * (512 / 16);           // 100000
    gemm_wmma_f16<<<blocks(tiles * 32, 128), 128, 0, stream>>>(h16, Wp1, bc1, QKVS1, NN, IN, 512);
  }
  // ---- conv1: edge attention ----
  edge_logits<<<blocks((long)NE*32, T), T, 0, stream>>>(ei, ea32, QKVS1, 512, 128, We1, W1, H1, a1, mx1);
  edge_soft<<<blocks((long)NE*H1, T), T, 0, stream>>>(ei, a1, mx1, sm1, H1);
  edge_msg<<<blocks((long)NE*32, T), T, 0, stream>>>(ei, ea32, QKVS1, 512, 256, We1, W1, H1, a1, sm1, agg1, W1);
  // ---- conv1: beta gate + BN1 + ELU ----
  beta_bn_elu1<<<blocks((long)NN*32, T), T, 0, stream>>>(agg1, QKVS1, Wb1, bn1_g, bn1_b, bn1_m, bn1_v, h1_16);

  // ---- conv2: projections via WMMA  [N,128] x [128,128] -> q|k|v|s ----
  {
    long tiles = (long)(NN / 16) * (128 / 16);           // 25000
    gemm_wmma_f16<<<blocks(tiles * 32, 128), 128, 0, stream>>>(h1_16, Wp2, bc2, QKVS2, NN, W1, 128);
  }
  // ---- conv2: edge attention (heads=1) ----
  edge_logits<<<blocks((long)NE*32, T), T, 0, stream>>>(ei, ea32, QKVS2, 128, 32, We2, HID, 1, a2, mx2);
  edge_soft<<<blocks((long)NE, T), T, 0, stream>>>(ei, a2, mx2, sm2, 1);
  edge_msg<<<blocks((long)NE*32, T), T, 0, stream>>>(ei, ea32, QKVS2, 128, 64, We2, HID, 1, a2, sm2, agg2, HID);
  // ---- conv2: beta gate + BN2 + ELU + pooling atomics ----
  beta_bn_elu2_pool<<<blocks((long)NN*32, T), T, 0, stream>>>(agg2, QKVS2, Wb2, bn2_g, bn2_b, bn2_m, bn2_v,
                                                             batch, psum, pmax, pcnt);
  // ---- final mean/max pooling -> [G, 64] ----
  pool_finalize<<<blocks(NG*64, T), T, 0, stream>>>(psum, pmax, pcnt, out);
}
